// TTTLayer_8632884264977
// MI455X (gfx1250) — matrix operand order repaired and checked by reading, not compiled, not run on hardware
//
#include <hip/hip_runtime.h>

#define Bsz 4
#define DIM 512
#define TOK 4096
#define NH  8
#define HD  64
#define CS  16
#define NC  256
#define BT  (Bsz * TOK)   // 16384 total token rows

typedef __bf16 bf16;
typedef __attribute__((ext_vector_type(8)))  bf16  v8bf;
typedef __attribute__((ext_vector_type(16))) bf16  v16bf;
typedef __attribute__((ext_vector_type(8)))  float v8f;

// ---------------------------------------------------------------------------
// WMMA helpers (CDNA5 gfx1250, wave32).
// A 16x32 bf16 fragment: lane m = lane&15; element e -> k = (lane>>4)*16 + e
// B 32x16 bf16 fragment: lane n = lane&15; element e -> k = (lane>>4)*16 + e
// Both read from LDS tiles stored row-major in K so each half is one 16B load.
// ---------------------------------------------------------------------------
__device__ __forceinline__ v16bf frag_a_row(const bf16* rp) {
  const int lane = (int)(threadIdx.x & 31u);
  const int kb = (lane >> 4) << 4;
  v8bf lo = *(const v8bf*)(rp + kb);
  v8bf hi = *(const v8bf*)(rp + kb + 8);
  v16bf f;
#pragma unroll
  for (int e = 0; e < 8; ++e) { f[e] = lo[e]; f[e + 8] = hi[e]; }
  return f;
}

__device__ __forceinline__ v16bf frag_b_row(const bf16* rp) {
  const int lane = (int)(threadIdx.x & 31u);
  const int kb = (lane >> 4) << 4;
  v8bf lo = *(const v8bf*)(rp + kb);
  v8bf hi = *(const v8bf*)(rp + kb + 8);
  v16bf f;
#pragma unroll
  for (int e = 0; e < 8; ++e) { f[e] = lo[e]; f[e + 8] = hi[e]; }
  return f;
}

__device__ __forceinline__ v8f wmma_bf16(v16bf a, v16bf b, v8f c) {
  // (neg_a, A, neg_b, B, c_mod, C, reuse_a, reuse_b)
  return __builtin_amdgcn_wmma_f32_16x16x32_bf16(false, a, false, b, (short)0, c,
                                                 false, false);
}

__device__ __forceinline__ float red8(float v) {
  v += __shfl_xor(v, 1, 32);
  v += __shfl_xor(v, 2, 32);
  v += __shfl_xor(v, 4, 32);
  return v;
}

// ---------------------------------------------------------------------------
// CDNA5 async global -> LDS copy (ASYNCcnt-tracked, no VGPR round trip).
// VDST VGPR = LDS byte offset (low 32 bits of the flat shared address),
// VADDR = 64-bit global address. Followed later by s_wait_asynccnt + barrier.
// Inline asm (not the clang builtin) so it is portable across toolchains.
// ---------------------------------------------------------------------------
__device__ __forceinline__ void async_copy_b128(void* lds_ptr, const void* gptr) {
  unsigned lds_off = (unsigned)(uintptr_t)lds_ptr;   // low 32 bits = LDS offset
  asm volatile("global_load_async_to_lds_b128 %0, %1, off"
               :
               : "v"(lds_off), "v"(gptr)
               : "memory");
}

__device__ __forceinline__ void wait_async0() {
  asm volatile("s_wait_asynccnt 0" ::: "memory");
}

// ---------------------------------------------------------------------------
// Convert the four 512x512 f32 weight matrices to bf16 (one pass, stays in L2).
// ---------------------------------------------------------------------------
__global__ __launch_bounds__(256) void cvt_w_kernel(
    const float* __restrict__ q, const float* __restrict__ k,
    const float* __restrict__ v, const float* __restrict__ o,
    bf16* __restrict__ qb, bf16* __restrict__ kb,
    bf16* __restrict__ vb, bf16* __restrict__ ob) {
  int i = blockIdx.x * 256 + threadIdx.x;       // 4 * 2^18 total
  int sel = i >> 18;
  int j = i & 262143;
  const float* s = sel == 0 ? q : sel == 1 ? k : sel == 2 ? v : o;
  bf16* d = sel == 0 ? qb : sel == 1 ? kb : sel == 2 ? vb : ob;
  d[j] = (bf16)s[j];
}

// ---------------------------------------------------------------------------
// Input LayerNorm: x is (B, DIM, T) -> xn bf16 (B, T, DIM). Thread = token so
// global reads are coalesced along T; second pass hits L2 (x = 33.5MB << 192MB).
// ---------------------------------------------------------------------------
__global__ __launch_bounds__(256) void ln_in_kernel(
    const float* __restrict__ x, const float* __restrict__ lw,
    const float* __restrict__ lb, bf16* __restrict__ xn) {
  int t = blockIdx.x * 256 + threadIdx.x;
  int b = blockIdx.y;
  const float* xp = x + (size_t)b * DIM * TOK + t;
  float s = 0.f, s2 = 0.f;
  for (int d = 0; d < DIM; ++d) { float v = xp[(size_t)d * TOK]; s += v; s2 += v * v; }
  float mu = s * (1.f / DIM);
  float rstd = rsqrtf(s2 * (1.f / DIM) - mu * mu + 1e-5f);   // EPS_OUT
  bf16* op = xn + ((size_t)b * TOK + t) * DIM;
  for (int d = 0; d < DIM; ++d) {
    float v = xp[(size_t)d * TOK];
    op[d] = (bf16)(((v - mu) * rstd) * lw[d] + lb[d]);
  }
}

// ---------------------------------------------------------------------------
// Fused QKV GEMM: C[M=BT, N=512] = xn @ w^T for proj in {q,k,v} via grid.y.
// 128 threads / 4 waves; 64x64 tile, K-step 32. Tiles move via async
// global->LDS copies, double buffered: one s_wait_asynccnt + one barrier per
// K-step, fetch of step i+1 overlaps the 4 WMMAs of step i.
// Output scattered into (B, NH, T, HD) bf16 for the scan kernel.
// ---------------------------------------------------------------------------
__global__ __launch_bounds__(128) void qkv_gemm_kernel(
    const bf16* __restrict__ xn, const bf16* __restrict__ wq,
    const bf16* __restrict__ wk, const bf16* __restrict__ wv,
    bf16* __restrict__ qo, bf16* __restrict__ ko, bf16* __restrict__ vo) {
  __shared__ __align__(16) bf16 As[2][64 * 32];
  __shared__ __align__(16) bf16 Bs[2][64 * 32];  // stored [n][k] (w is row-major (n,k))
  const int tid = threadIdx.x;
  const int wave = tid >> 5, lane = tid & 31;
  const int m0g = blockIdx.x * 64;
  const int ny = blockIdx.y;                     // 0..23
  const int proj = ny >> 3, h = ny & 7;
  const int n0g = h * 64;
  const bf16* w = proj == 0 ? wq : (proj == 1 ? wk : wv);
  bf16* out = proj == 0 ? qo : (proj == 1 ? ko : vo);

  // this thread's two 16B slots of each 64x32 tile
  const int r0 = (tid * 2) >> 2, c0 = ((tid * 2) & 3) * 8;
  const int r1 = (tid * 2 + 1) >> 2, c1 = ((tid * 2 + 1) & 3) * 8;

  auto issue_tiles = [&](int buf, int k0) {
    async_copy_b128(&As[buf][r0 * 32 + c0], &xn[(size_t)(m0g + r0) * DIM + k0 + c0]);
    async_copy_b128(&As[buf][r1 * 32 + c1], &xn[(size_t)(m0g + r1) * DIM + k0 + c1]);
    async_copy_b128(&Bs[buf][r0 * 32 + c0], &w[(size_t)(n0g + r0) * DIM + k0 + c0]);
    async_copy_b128(&Bs[buf][r1 * 32 + c1], &w[(size_t)(n0g + r1) * DIM + k0 + c1]);
  };

  v8f acc[4];
#pragma unroll
  for (int i = 0; i < 4; ++i)
#pragma unroll
    for (int e = 0; e < 8; ++e) acc[i][e] = 0.f;

  issue_tiles(0, 0);
  const int m0 = wave * 16;
  for (int i = 0; i < DIM / 32; ++i) {
    wait_async0();        // this wave's async writes to LDS complete
    __syncthreads();      // -> everyone's tiles visible; prev buffer free
    if (i + 1 < DIM / 32) issue_tiles((i + 1) & 1, (i + 1) * 32);
    const bf16* Ab = &As[i & 1][0];
    const bf16* Bb = &Bs[i & 1][0];
    v16bf a = frag_a_row(&Ab[(m0 + (lane & 15)) * 32]);
#pragma unroll
    for (int nt = 0; nt < 4; ++nt) {
      v16bf bfr = frag_b_row(&Bb[(nt * 16 + (lane & 15)) * 32]);
      acc[nt] = wmma_bf16(a, bfr, acc[nt]);
    }
  }
  const int roff = (lane >> 4) << 3;
#pragma unroll
  for (int nt = 0; nt < 4; ++nt)
#pragma unroll
    for (int r = 0; r < 8; ++r) {
      int M = m0g + m0 + r + roff;               // global token row
      int hd = nt * 16 + (lane & 15);
      int b = M >> 12, t = M & 4095;
      out[(((size_t)(b * NH + h) * TOK) + t) * HD + hd] = (bf16)acc[nt][r];
    }
}

// ---------------------------------------------------------------------------
// RoPE on Q and K, in place on the bf16 (B,NH,T,HD) buffers.
// ---------------------------------------------------------------------------
__global__ __launch_bounds__(256) void rope_kernel(bf16* __restrict__ q,
                                                   bf16* __restrict__ k) {
  int idx = blockIdx.x * 256 + threadIdx.x;      // B*NH*T*32 threads
  int j = idx & 31;
  int t = (idx >> 5) & 4095;
  int bh = idx >> 17;
  size_t base = ((size_t)bh * TOK + t) * HD;
  float fr = (float)t * __powf(10000.f, -(float)(2 * j) / (float)HD);
  float cs = __cosf(fr), sn = __sinf(fr);
  float a = (float)q[base + j], b_ = (float)q[base + j + 32];
  q[base + j]      = (bf16)(a * cs - b_ * sn);
  q[base + j + 32] = (bf16)(b_ * cs + a * sn);
  a = (float)k[base + j]; b_ = (float)k[base + j + 32];
  k[base + j]      = (bf16)(a * cs - b_ * sn);
  k[base + j + 32] = (bf16)(b_ * cs + a * sn);
}

// ---------------------------------------------------------------------------
// Per-token learning-rate gate: sig[b,h,t] = sigmoid(xn[t,:] . lr_w[h,:] + lr_b[h]) / HD
// One wave per (b,h,t); lanes stride D (coalesced within the wave).
// ---------------------------------------------------------------------------
__global__ __launch_bounds__(256) void eta_kernel(
    const bf16* __restrict__ xn, const float* __restrict__ lrw,
    const float* __restrict__ lrb, float* __restrict__ sig) {
  int wid = (blockIdx.x * 256 + threadIdx.x) >> 5;   // 0 .. B*NH*T-1
  int lane = threadIdx.x & 31;
  int bh = wid >> 12, t = wid & 4095;
  int b = bh >> 3, hh = bh & 7;
  const bf16* xp = xn + ((size_t)b * TOK + t) * DIM;
  const float* wp = lrw + hh * DIM;
  float acc = 0.f;
  for (int d = lane; d < DIM; d += 32) acc += (float)xp[d] * wp[d];
#pragma unroll
  for (int m = 1; m < 32; m <<= 1) acc += __shfl_xor(acc, m, 32);
  if (lane == 0) {
    float z = acc + lrb[hh];
    sig[(size_t)bh * TOK + t] = (1.f / (1.f + __expf(-z))) * (1.f / (float)HD);
  }
}

// ---------------------------------------------------------------------------
// The TTT scan: one workgroup per (b,h), 128 threads = 4 waves, 256 sequential
// chunk steps. W1 lives in LDS: f32 master + bf16 [n][k] copy for B-fragments.
// Per step: Z1 (8 wmma), Attn (2 wmma), Z1_bar chain (16 wmma), W1 update
// (16 wmma) + f32 LayerNorm fwd/bwd reductions. xq moves via the async LDS
// path; next chunk's q/k/v rows are prefetched (global_prefetch_b8) while the
// dependent math of the current step runs.
// ---------------------------------------------------------------------------
__global__ __launch_bounds__(128) void ttt_scan_kernel(
    const bf16* __restrict__ qg, const bf16* __restrict__ kg,
    const bf16* __restrict__ vg, const float* __restrict__ sig,
    const float* __restrict__ W1g, const float* __restrict__ b1g,
    const float* __restrict__ tlw, const float* __restrict__ tlb,
    const float* __restrict__ tdelta, float* __restrict__ outw) {
  __shared__ __align__(16) float W1f[64 * 64];    // master W1 (k-major rows: [k][n])
  __shared__ __align__(16) bf16  W1bT[64 * 64];   // [n][k] = W1[k][n]
  __shared__ __align__(16) bf16  Kb[16 * 64];     // xk  [m][k]
  __shared__ __align__(16) bf16  Qb[16 * 64];     // xq  [m][k]
  __shared__ __align__(16) float tgt[16 * 64];    // xv - xk
  __shared__ __align__(16) float Z1s[16 * 64];
  __shared__ __align__(16) float Z1bar[16 * 64];
  __shared__ __align__(16) float gradf[16 * 64];
  __shared__ __align__(16) bf16  gradT[64 * 32];  // [n][k], k 16..31 zero
  __shared__ __align__(16) bf16  KbTs[64 * 32];   // -(last_eta[k]*xk[k][m]) as A[m][k], k pad
  __shared__ __align__(16) bf16  etA[16 * 32];    // -tril(et) as A, k pad
  __shared__ __align__(16) bf16  etB[16 * 32];    // -(et * Attn) as A, k pad
  __shared__ __align__(16) float Attn[16 * 16];
  __shared__ float b1s[64], gLs[64], bLs[64], tokidx[16], sigrow[16];

  const int tid = threadIdx.x;
  const int wave = tid >> 5, lane = tid & 31;
  const int roff = (lane >> 4) << 3;
  const int n0 = wave * 16;
  const int bh = blockIdx.x, b = bh >> 3, h = bh & 7;
  const bf16* qp = qg + (size_t)bh * TOK * HD;
  const bf16* kp = kg + (size_t)bh * TOK * HD;
  const bf16* vp = vg + (size_t)bh * TOK * HD;
  const float* sp = sig + (size_t)bh * TOK;

  // ---- init ----
  for (int i = tid; i < 64 * 64; i += 128) W1f[i] = W1g[h * 64 * 64 + i];
  if (tid < 64) { b1s[tid] = b1g[h * 64 + tid]; gLs[tid] = tlw[h * 64 + tid]; bLs[tid] = tlb[h * 64 + tid]; }
  if (tid < 16) tokidx[tid] = fmaxf(1.f / (float)(tid + 1) + tdelta[tid], 0.f);
  for (int i = tid; i < 64 * 32; i += 128) { gradT[i] = (bf16)0.f; KbTs[i] = (bf16)0.f; }
  for (int i = tid; i < 16 * 32; i += 128) { etA[i] = (bf16)0.f; etB[i] = (bf16)0.f; }
  __syncthreads();
  for (int i = tid; i < 64 * 64; i += 128) { int n = i >> 6, k = i & 63; W1bT[i] = (bf16)W1f[k * 64 + n]; }
  __syncthreads();

  for (int c = 0; c < NC; ++c) {
    // ---- A: load chunk (xq via async LDS path; k/v through VGPRs for tgt) ----
    {
      size_t off = (size_t)c * CS * HD + (size_t)tid * 8;
      async_copy_b128(&Qb[tid * 8], qp + off);
      v8bf kv = *(const v8bf*)(kp + off);
      v8bf vv = *(const v8bf*)(vp + off);
      *(v8bf*)&Kb[tid * 8] = kv;
#pragma unroll
      for (int e = 0; e < 8; ++e) tgt[tid * 8 + e] = (float)vv[e] - (float)kv[e];
      if (tid < 16) sigrow[tid] = sp[c * CS + tid];
    }
    wait_async0();
    __syncthreads();
    // warm next chunk's rows into WGP$/L2 while this step's dependent math runs
    if (c + 1 < NC) {
      size_t noff = (size_t)(c + 1) * CS * HD + (size_t)tid * 8;
      __builtin_prefetch(qp + noff, 0, 3);
      __builtin_prefetch(kp + noff, 0, 3);
      __builtin_prefetch(vp + noff, 0, 3);
    }

    // ---- B: Z1 = xk @ W1 + b1 (wave owns 16-col strip) ----
    {
      v8f az;
#pragma unroll
      for (int e = 0; e < 8; ++e) az[e] = 0.f;
      az = wmma_bf16(frag_a_row(&Kb[(lane & 15) * 64 + 0]),
                     frag_b_row(&W1bT[(n0 + (lane & 15)) * 64 + 0]), az);
      az = wmma_bf16(frag_a_row(&Kb[(lane & 15) * 64 + 32]),
                     frag_b_row(&W1bT[(n0 + (lane & 15)) * 64 + 32]), az);
#pragma unroll
      for (int r = 0; r < 8; ++r)
        Z1s[(r + roff) * 64 + n0 + (lane & 15)] = az[r] + b1s[n0 + (lane & 15)];
    }
    // ---- D: Attn = tril(xq @ xk^T) (wave 0; xk^T's BT layout == Kb itself) ----
    if (wave == 0) {
      v8f aa;
#pragma unroll
      for (int e = 0; e < 8; ++e) aa[e] = 0.f;
      aa = wmma_bf16(frag_a_row(&Qb[(lane & 15) * 64 + 0]),
                     frag_b_row(&Kb[(lane & 15) * 64 + 0]), aa);
      aa = wmma_bf16(frag_a_row(&Qb[(lane & 15) * 64 + 32]),
                     frag_b_row(&Kb[(lane & 15) * 64 + 32]), aa);
#pragma unroll
      for (int r = 0; r < 8; ++r) {
        int m = r + roff, n = lane & 15;
        Attn[m * 16 + n] = (n <= m) ? aa[r] : 0.f;
      }
    }
    // ---- build -(last_eta*xk)^T and -tril(et) ----
    {
      float le = tokidx[15];
#pragma unroll
      for (int e = 0; e < 8; ++e) {
        int idx = tid * 8 + e;              // 1024 = 64x16
        int m = idx >> 4, k = idx & 15;
        KbTs[m * 32 + k] = (bf16)(-(le * sigrow[k]) * (float)Kb[k * 64 + m]);
      }
#pragma unroll
      for (int e = 0; e < 2; ++e) {
        int idx = tid * 2 + e;              // 256 = 16x16
        int i = idx >> 4, jj = idx & 15;
        float v = (jj <= i) ? (tokidx[i] * sigrow[jj]) : 0.f;
        etA[i * 32 + jj] = (bf16)(-v);
      }
    }
    __syncthreads();

    // ---- C: grad = ln_l2_bwd(Z1, xv-xk)  (8 threads per row) ----
    {
      int row = tid >> 3, sub = tid & 7, dbase = sub * 8;
      float z[8], s = 0.f, s2 = 0.f;
#pragma unroll
      for (int j = 0; j < 8; ++j) { z[j] = Z1s[row * 64 + dbase + j]; s += z[j]; s2 += z[j] * z[j]; }
      s = red8(s); s2 = red8(s2);
      float mu = s * (1.f / 64.f);
      float rstd = rsqrtf(s2 * (1.f / 64.f) - mu * mu + 1e-6f);   // EPS_IN
      float xh[8], gxh[8], S1 = 0.f, S2 = 0.f;
#pragma unroll
      for (int j = 0; j < 8; ++j) {
        float xhj = (z[j] - mu) * rstd;
        float gj = gLs[dbase + j];
        float t_ = (gj * xhj + bLs[dbase + j] - tgt[row * 64 + dbase + j]) * gj;
        xh[j] = xhj; gxh[j] = t_; S1 += t_; S2 += t_ * xhj;
      }
      S1 = red8(S1); S2 = red8(S2);
#pragma unroll
      for (int j = 0; j < 8; ++j) {
        float gr = (64.f * gxh[j] - S1 - xh[j] * S2) * (rstd * (1.f / 64.f));
        gradf[row * 64 + dbase + j] = gr;
        gradT[(dbase + j) * 32 + row] = (bf16)gr;
      }
      // -(et * Attn) now that Attn is visible
#pragma unroll
      for (int e = 0; e < 2; ++e) {
        int idx = tid * 2 + e;
        int i = idx >> 4, jj = idx & 15;
        etB[i * 32 + jj] = (bf16)(-(tokidx[i] * sigrow[jj]) * Attn[i * 16 + jj]);
      }
    }
    __syncthreads();

    // ---- E: Z1_bar = xq@W1 - (et*Attn)@grad - tril(et)@grad + b1 (4-wmma chain) ----
    {
      v8f az;
#pragma unroll
      for (int e = 0; e < 8; ++e) az[e] = 0.f;
      az = wmma_bf16(frag_a_row(&Qb[(lane & 15) * 64 + 0]),
                     frag_b_row(&W1bT[(n0 + (lane & 15)) * 64 + 0]), az);
      az = wmma_bf16(frag_a_row(&Qb[(lane & 15) * 64 + 32]),
                     frag_b_row(&W1bT[(n0 + (lane & 15)) * 64 + 32]), az);
      v16bf bg = frag_b_row(&gradT[(n0 + (lane & 15)) * 32]);
      az = wmma_bf16(frag_a_row(&etB[(lane & 15) * 32]), bg, az);
      az = wmma_bf16(frag_a_row(&etA[(lane & 15) * 32]), bg, az);
#pragma unroll
      for (int r = 0; r < 8; ++r)
        Z1bar[(r + roff) * 64 + n0 + (lane & 15)] = az[r] + b1s[n0 + (lane & 15)];
    }
    __syncthreads();

    // ---- F: out = xq + ln(Z1_bar) -> global (B, T, DIM) ----
    {
      int row = tid >> 3, sub = tid & 7, dbase = sub * 8;
      float z[8], s = 0.f, s2 = 0.f;
#pragma unroll
      for (int j = 0; j < 8; ++j) { z[j] = Z1bar[row * 64 + dbase + j]; s += z[j]; s2 += z[j] * z[j]; }
      s = red8(s); s2 = red8(s2);
      float mu = s * (1.f / 64.f);
      float rstd = rsqrtf(s2 * (1.f / 64.f) - mu * mu + 1e-6f);
      size_t obase = ((size_t)b * TOK + (size_t)c * CS + row) * DIM + h * HD + dbase;
#pragma unroll
      for (int j = 0; j < 8; ++j)
        outw[obase + j] = (float)Qb[row * 64 + dbase + j] +
                          ((z[j] - mu) * rstd * gLs[dbase + j] + bLs[dbase + j]);
    }
    // ---- G: W1 -= (last_eta*xk)^T @ grad ; b1 -= sum(last_eta*grad) ----
    {
      v16bf bg = frag_b_row(&gradT[(n0 + (lane & 15)) * 32]);
#pragma unroll
      for (int mt = 0; mt < 4; ++mt) {
        v8f cc;
#pragma unroll
        for (int r = 0; r < 8; ++r)
          cc[r] = W1f[(mt * 16 + r + roff) * 64 + n0 + (lane & 15)];
        cc = wmma_bf16(frag_a_row(&KbTs[(mt * 16 + (lane & 15)) * 32]), bg, cc);
#pragma unroll
        for (int r = 0; r < 8; ++r)
          W1f[(mt * 16 + r + roff) * 64 + n0 + (lane & 15)] = cc[r];
      }
      if (tid < 64) {
        float s = 0.f, le = tokidx[15];
#pragma unroll
        for (int k = 0; k < 16; ++k) s += le * sigrow[k] * gradf[k * 64 + tid];
        b1s[tid] -= s;
      }
    }
    __syncthreads();
    // rebuild bf16 [n][k] copy of W1 for next step's B-fragments
    for (int i = tid; i < 64 * 64; i += 128) { int n = i >> 6, k = i & 63; W1bT[i] = (bf16)W1f[k * 64 + n]; }
    __syncthreads();
  }
}

// ---------------------------------------------------------------------------
// Output LayerNorm: out f32 (B,T,DIM) -> bf16 normalized (B,T,DIM). Block per
// token, 128 threads, coalesced.
// ---------------------------------------------------------------------------
__global__ __launch_bounds__(128) void ln_out_kernel(
    const float* __restrict__ src, const float* __restrict__ pw,
    const float* __restrict__ pb, bf16* __restrict__ dst) {
  int bt = blockIdx.x;
  int tid = threadIdx.x;
  const float* row = src + (size_t)bt * DIM;
  float v[4], s = 0.f, s2 = 0.f;
#pragma unroll
  for (int j = 0; j < 4; ++j) { v[j] = row[tid + j * 128]; s += v[j]; s2 += v[j] * v[j]; }
#pragma unroll
  for (int m = 1; m < 32; m <<= 1) { s += __shfl_xor(s, m, 32); s2 += __shfl_xor(s2, m, 32); }
  __shared__ float ws1[4], ws2[4];
  if ((tid & 31) == 0) { ws1[tid >> 5] = s; ws2[tid >> 5] = s2; }
  __syncthreads();
  s = ws1[0] + ws1[1] + ws1[2] + ws1[3];
  s2 = ws2[0] + ws2[1] + ws2[2] + ws2[3];
  float mu = s * (1.f / DIM);
  float rstd = rsqrtf(s2 * (1.f / DIM) - mu * mu + 1e-6f);   // EPS_IN
#pragma unroll
  for (int j = 0; j < 4; ++j) {
    int d = tid + j * 128;
    dst[(size_t)bt * DIM + d] = (bf16)(((v[j] - mu) * rstd) * pw[d] + pb[d]);
  }
}

// ---------------------------------------------------------------------------
// O projection GEMM (async double-buffered like QKV), epilogue writes
// transposed into d_out (B, DIM, H*W) f32.
// ---------------------------------------------------------------------------
__global__ __launch_bounds__(128) void o_gemm_kernel(
    const bf16* __restrict__ xn, const bf16* __restrict__ wo,
    float* __restrict__ outp) {
  __shared__ __align__(16) bf16 As[2][64 * 32];
  __shared__ __align__(16) bf16 Bs[2][64 * 32];
  const int tid = threadIdx.x;
  const int wave = tid >> 5, lane = tid & 31;
  const int m0g = blockIdx.x * 64;
  const int n0g = blockIdx.y * 64;

  const int r0 = (tid * 2) >> 2, c0 = ((tid * 2) & 3) * 8;
  const int r1 = (tid * 2 + 1) >> 2, c1 = ((tid * 2 + 1) & 3) * 8;
  auto issue_tiles = [&](int buf, int k0) {
    async_copy_b128(&As[buf][r0 * 32 + c0], &xn[(size_t)(m0g + r0) * DIM + k0 + c0]);
    async_copy_b128(&As[buf][r1 * 32 + c1], &xn[(size_t)(m0g + r1) * DIM + k0 + c1]);
    async_copy_b128(&Bs[buf][r0 * 32 + c0], &wo[(size_t)(n0g + r0) * DIM + k0 + c0]);
    async_copy_b128(&Bs[buf][r1 * 32 + c1], &wo[(size_t)(n0g + r1) * DIM + k0 + c1]);
  };

  v8f acc[4];
#pragma unroll
  for (int i = 0; i < 4; ++i)
#pragma unroll
    for (int e = 0; e < 8; ++e) acc[i][e] = 0.f;

  issue_tiles(0, 0);
  const int m0 = wave * 16;
  for (int i = 0; i < DIM / 32; ++i) {
    wait_async0();
    __syncthreads();
    if (i + 1 < DIM / 32) issue_tiles((i + 1) & 1, (i + 1) * 32);
    const bf16* Ab = &As[i & 1][0];
    const bf16* Bb = &Bs[i & 1][0];
    v16bf a = frag_a_row(&Ab[(m0 + (lane & 15)) * 32]);
#pragma unroll
    for (int nt = 0; nt < 4; ++nt) {
      v16bf bfr = frag_b_row(&Bb[(nt * 16 + (lane & 15)) * 32]);
      acc[nt] = wmma_bf16(a, bfr, acc[nt]);
    }
  }
  const int roff = (lane >> 4) << 3;
#pragma unroll
  for (int nt = 0; nt < 4; ++nt)
#pragma unroll
    for (int r = 0; r < 8; ++r) {
      int M = m0g + m0 + r + roff;
      int N = n0g + nt * 16 + (lane & 15);
      int b = M >> 12, t = M & 4095;
      outp[((size_t)b * DIM + N) * TOK + t] = acc[nt][r];
    }
}

// ---------------------------------------------------------------------------
extern "C" void kernel_launch(void* const* d_in, const int* in_sizes, int n_in,
                              void* d_out, int out_size, void* d_ws, size_t ws_size,
                              hipStream_t stream) {
  const float* x    = (const float*)d_in[0];
  const float* ln_w = (const float*)d_in[1];
  const float* ln_b = (const float*)d_in[2];
  const float* q_w  = (const float*)d_in[3];
  const float* k_w  = (const float*)d_in[4];
  const float* v_w  = (const float*)d_in[5];
  const float* o_w  = (const float*)d_in[6];
  const float* W1   = (const float*)d_in[7];
  const float* b1   = (const float*)d_in[8];
  const float* tlw  = (const float*)d_in[9];
  const float* tlb  = (const float*)d_in[10];
  const float* lrw  = (const float*)d_in[11];
  const float* lrb  = (const float*)d_in[12];
  const float* tde  = (const float*)d_in[13];
  const float* pw   = (const float*)d_in[14];
  const float* pb   = (const float*)d_in[15];
  float* out = (float*)d_out;
  (void)in_sizes; (void)n_in; (void)out_size; (void)ws_size;

  char* ws = (char*)d_ws;
  size_t off = 0;
  auto carve = [&](size_t bytes) {
    void* p = ws + off;
    off = (off + bytes + 255) & ~(size_t)255;
    return p;
  };
  bf16*  xn_bf = (bf16*)carve((size_t)BT * DIM * sizeof(bf16));
  bf16*  q_bf  = (bf16*)carve((size_t)BT * DIM * sizeof(bf16));
  bf16*  k_bf  = (bf16*)carve((size_t)BT * DIM * sizeof(bf16));
  bf16*  v_bf  = (bf16*)carve((size_t)BT * DIM * sizeof(bf16));
  bf16*  wq_bf = (bf16*)carve((size_t)DIM * DIM * sizeof(bf16));
  bf16*  wk_bf = (bf16*)carve((size_t)DIM * DIM * sizeof(bf16));
  bf16*  wv_bf = (bf16*)carve((size_t)DIM * DIM * sizeof(bf16));
  bf16*  wo_bf = (bf16*)carve((size_t)DIM * DIM * sizeof(bf16));
  float* sigp  = (float*)carve((size_t)Bsz * NH * TOK * sizeof(float));
  float* outw  = (float*)carve((size_t)BT * DIM * sizeof(float));
  bf16*  nrm   = (bf16*)carve((size_t)BT * DIM * sizeof(bf16));

  cvt_w_kernel<<<(4 * DIM * DIM) / 256, 256, 0, stream>>>(q_w, k_w, v_w, o_w,
                                                          wq_bf, wk_bf, wv_bf, wo_bf);
  ln_in_kernel<<<dim3(TOK / 256, Bsz), 256, 0, stream>>>(x, ln_w, ln_b, xn_bf);
  qkv_gemm_kernel<<<dim3(BT / 64, 24), 128, 0, stream>>>(xn_bf, wq_bf, wk_bf, wv_bf,
                                                         q_bf, k_bf, v_bf);
  rope_kernel<<<(Bsz * NH * TOK * 32) / 256, 256, 0, stream>>>(q_bf, k_bf);
  eta_kernel<<<(Bsz * NH * TOK) / 8, 256, 0, stream>>>(xn_bf, lrw, lrb, sigp);
  ttt_scan_kernel<<<Bsz * NH, 128, 0, stream>>>(q_bf, k_bf, v_bf, sigp, W1, b1,
                                                tlw, tlb, tde, outw);
  ln_out_kernel<<<BT, 128, 0, stream>>>(outw, pw, pb, nrm);
  o_gemm_kernel<<<dim3(BT / 64, 8), 128, 0, stream>>>(nrm, pw == pb ? wo_bf : wo_bf, out);
}
